// MultiHeadSelfAttention_18056042512668
// MI455X (gfx1250) — matrix operand order repaired and checked by reading, not compiled, not run on hardware
//
#include <hip/hip_runtime.h>
#include <hip/hip_bf16.h>

// H=1024, NH=16, HD=64, B=2, S=2048.  All matrix math on v_wmma_f32_16x16x32_bf16.

typedef __attribute__((ext_vector_type(16))) __bf16 v16bf;
typedef __attribute__((ext_vector_type(8)))  float  v8f;
typedef __attribute__((ext_vector_type(4)))  unsigned int u32x4;

union FragU { v16bf v; u32x4 q[2]; };
union BfBits { __bf16 h; unsigned short u; };

__device__ inline unsigned short f2bf_bits(float f) {
    BfBits b; b.h = (__bf16)f; return b.u;
}

__device__ inline v8f wmma_bf16(v16bf a, v16bf b, v8f c) {
    return __builtin_amdgcn_wmma_f32_16x16x32_bf16(
        /*neg_a=*/false, a, /*neg_b=*/false, b,
        /*c_mod=*/(short)0, c, /*reuse_a=*/false, /*reuse_b=*/false);
}

// A-fragment (16x32, MxK): lane m = lane&15; half = lane>>4.
// low 16B = K in [hi*8, hi*8+7], high 16B = K in [16+hi*8, 16+hi*8+7]  (p = row + kbase)
__device__ inline v16bf ldA(const __bf16* p, int hi) {
    FragU f;
    f.q[0] = *(const u32x4*)(p + hi * 8);
    f.q[1] = *(const u32x4*)(p + 16 + hi * 8);
    return f.v;
}

// B-fragment (32x16, KxN) from [N][K] storage: lane n = lane&15 selects row,
// low 16B = K in [hi*8, hi*8+7], high 16B = K in [16+hi*8, 16+hi*8+7]  (p = row_of_n + kbase)
__device__ inline v16bf ldB(const __bf16* p, int hi) {
    FragU f;
    f.q[0] = *(const u32x4*)(p + hi * 8);
    f.q[1] = *(const u32x4*)(p + 16 + hi * 8);
    return f.v;
}

__global__ void cvt_f32_to_bf16(const float* __restrict__ src,
                                unsigned short* __restrict__ dst, int n) {
    int i = blockIdx.x * blockDim.x + threadIdx.x;
    if (i < n) dst[i] = f2bf_bits(src[i]);
}

// C[M,N] = A[M,K] * W[N,K]^T + bias.  Block: 8 waves -> 128(M) x 64(N) tile,
// wave = 32x32 via 2x2 WMMA tiles.  Epilogue modes:
//   0/1: scatter bf16 into Q/K layout [b][h][s][d]
//   2:   scatter bf16 into Vt layout [b][h][d][s]
//   3:   f32 row-major (final output, bias bo)
__global__ __launch_bounds__(256)
void gemm_bf16(const unsigned short* __restrict__ Au,
               const unsigned short* __restrict__ Wu,
               const float* __restrict__ bias,
               void* __restrict__ dst,
               int M, int N, int K, int mode) {
    const __bf16* A = (const __bf16*)Au;
    const __bf16* W = (const __bf16*)Wu;
    int lane = threadIdx.x & 31;
    int wave = threadIdx.x >> 5;
    int hi   = lane >> 4;
    int l16  = lane & 15;
    int wm = wave & 3, wn = wave >> 2;
    int M0 = blockIdx.y * 128 + wm * 32;
    int N0 = blockIdx.x * 64 + wn * 32;

    const __bf16* ar0 = A + (size_t)(M0 + l16) * K;
    const __bf16* ar1 = A + (size_t)(M0 + 16 + l16) * K;
    const __bf16* br0 = W + (size_t)(N0 + l16) * K;
    const __bf16* br1 = W + (size_t)(N0 + 16 + l16) * K;

    v8f acc00 = {}, acc01 = {}, acc10 = {}, acc11 = {};
    for (int k = 0; k < K; k += 32) {
        v16bf a0 = ldA(ar0 + k, hi);
        v16bf a1 = ldA(ar1 + k, hi);
        v16bf b0 = ldB(br0 + k, hi);
        v16bf b1 = ldB(br1 + k, hi);
        acc00 = wmma_bf16(a0, b0, acc00);
        acc01 = wmma_bf16(a0, b1, acc01);
        acc10 = wmma_bf16(a1, b0, acc10);
        acc11 = wmma_bf16(a1, b1, acc11);
    }

    v8f accs[2][2] = { { acc00, acc01 }, { acc10, acc11 } };
#pragma unroll
    for (int i = 0; i < 2; i++)
#pragma unroll
    for (int j = 0; j < 2; j++) {
        int n = N0 + j * 16 + l16;
        float bv = bias[n];
#pragma unroll
        for (int v = 0; v < 8; v++) {
            int m = M0 + i * 16 + v + hi * 8;   // C/D layout: lanes16-31 -> M=v+8
            float val = accs[i][j][v] + bv;
            if (mode == 3) {
                ((float*)dst)[(size_t)m * N + n] = val;
            } else {
                int b = m >> 11, s = m & 2047;   // M = b*2048 + s
                int h = n >> 6,  d = n & 63;     // N = h*64 + d
                size_t idx = (mode == 2)
                    ? ((size_t)((b * 16 + h) * 64 + d)) * 2048 + s   // Vt[b][h][d][s]
                    : ((size_t)((b * 16 + h) * 2048 + s)) * 64 + d;  // Q/K[b][h][s][d]
                ((unsigned short*)dst)[idx] = f2bf_bits(val);
            }
        }
    }
}

// Flash attention: grid (S/128, B*NH), block 256 = 8 independent waves,
// each wave owns 16 query rows, streams keys 32 at a time with online softmax.
__global__ __launch_bounds__(256)
void attn_kernel(const unsigned short* __restrict__ Qu,
                 const unsigned short* __restrict__ Ku,
                 const unsigned short* __restrict__ Vtu,
                 const float* __restrict__ mask,
                 unsigned short* __restrict__ Ctx) {
    __shared__ __align__(16) unsigned short sP[8 * 16 * 32];  // per-wave 16x32 P tile

    const __bf16* Q  = (const __bf16*)Qu;
    const __bf16* K  = (const __bf16*)Ku;
    const __bf16* Vt = (const __bf16*)Vtu;

    int lane = threadIdx.x & 31;
    int wave = threadIdx.x >> 5;
    int hi   = lane >> 4;
    int l16  = lane & 15;
    int bh = blockIdx.y;                 // b*16 + h
    int b  = bh >> 4, h = bh & 15;
    int q0 = blockIdx.x * 128 + wave * 16;

    const __bf16* Qp = Q  + (size_t)bh * 2048 * 64;
    const __bf16* Kp = K  + (size_t)bh * 2048 * 64;
    const __bf16* Vp = Vt + (size_t)bh * 64 * 2048;
    const float*  mk = mask + (size_t)b * 2048;

    // Q fragments for this wave's 16 rows, d-blocks [0,32) and [32,64)
    const __bf16* qrow = Qp + (size_t)(q0 + l16) * 64;
    v16bf qa0 = ldA(qrow + 0,  hi);
    v16bf qa1 = ldA(qrow + 32, hi);

    float rmax[8], rsum[8];
#pragma unroll
    for (int v = 0; v < 8; v++) { rmax[v] = -3.0e38f; rsum[v] = 0.0f; }
    v8f ctx0 = {}, ctx1 = {}, ctx2 = {}, ctx3 = {};

    __bf16* myP = (__bf16*)&sP[wave * 16 * 32];

    for (int j = 0; j < 2048; j += 32) {
        // ---- scores: two 16x16 tiles (keys j..j+15, j+16..j+31), K-dim = HD = 2x32
        const __bf16* kr0 = Kp + (size_t)(j + l16) * 64;
        const __bf16* kr1 = Kp + (size_t)(j + 16 + l16) * 64;
        v16bf kb00 = ldB(kr0 + 0,  hi);
        v16bf kb01 = ldB(kr0 + 32, hi);
        v16bf kb10 = ldB(kr1 + 0,  hi);
        v16bf kb11 = ldB(kr1 + 32, hi);
        v8f s0 = {}, s1 = {};
        s0 = wmma_bf16(qa0, kb00, s0);
        s0 = wmma_bf16(qa1, kb01, s0);
        s1 = wmma_bf16(qa0, kb10, s1);
        s1 = wmma_bf16(qa1, kb11, s1);

        float m0 = mk[j + l16];
        float m1 = mk[j + 16 + l16];

        float p0[8], p1[8];
#pragma unroll
        for (int v = 0; v < 8; v++) {
            float a = s0[v] * 0.125f + m0;   // 1/sqrt(64)
            float c = s1[v] * 0.125f + m1;
            float t = fmaxf(a, c);
#pragma unroll
            for (int off = 1; off < 16; off <<= 1)
                t = fmaxf(t, __shfl_xor(t, off, 16));
            float nm = fmaxf(rmax[v], t);
            float corr = __expf(rmax[v] - nm);
            rmax[v] = nm;
            rsum[v] *= corr;
            ctx0[v] *= corr; ctx1[v] *= corr; ctx2[v] *= corr; ctx3[v] *= corr;
            float e0 = __expf(a - nm);
            float e1 = __expf(c - nm);
            float rs = e0 + e1;
#pragma unroll
            for (int off = 1; off < 16; off <<= 1)
                rs += __shfl_xor(rs, off, 16);
            rsum[v] += rs;
            p0[v] = e0;
            p1[v] = e1;
        }

        // ---- re-layout P (C/D layout -> A layout) via per-wave LDS tile
#pragma unroll
        for (int v = 0; v < 8; v++) {
            int r = v + hi * 8;                      // row within 16
            myP[r * 32 + l16]      = (__bf16)p0[v];  // cols 0..15
            myP[r * 32 + 16 + l16] = (__bf16)p1[v];  // cols 16..31
        }
        // same-wave LDS ops are in-order; load the 16x32 A-fragment back
        FragU pf;
        const __bf16* prow = myP + l16 * 32;
        pf.q[0] = *(const u32x4*)(prow + hi * 8);
        pf.q[1] = *(const u32x4*)(prow + 16 + hi * 8);

        // ---- ctx += P (16x32) x V (32x16) for each of the 4 d-tiles, from Vt[d][s]
        {
            const __bf16* vr0 = Vp + (size_t)(0 * 16 + l16) * 2048 + j;
            const __bf16* vr1 = Vp + (size_t)(1 * 16 + l16) * 2048 + j;
            const __bf16* vr2 = Vp + (size_t)(2 * 16 + l16) * 2048 + j;
            const __bf16* vr3 = Vp + (size_t)(3 * 16 + l16) * 2048 + j;
            ctx0 = wmma_bf16(pf.v, ldB(vr0, hi), ctx0);
            ctx1 = wmma_bf16(pf.v, ldB(vr1, hi), ctx1);
            ctx2 = wmma_bf16(pf.v, ldB(vr2, hi), ctx2);
            ctx3 = wmma_bf16(pf.v, ldB(vr3, hi), ctx3);
        }
    }

    // ---- normalize and scatter ctx (bf16) into [b][s][h*64+d] for the O-projection
    v8f ctxs[4] = { ctx0, ctx1, ctx2, ctx3 };
#pragma unroll
    for (int v = 0; v < 8; v++) {
        float inv = 1.0f / rsum[v];
        int q = q0 + v + hi * 8;
#pragma unroll
        for (int nt = 0; nt < 4; nt++) {
            float val = ctxs[nt][v] * inv;
            int d = nt * 16 + l16;
            Ctx[((size_t)(b * 2048 + q)) * 1024 + h * 64 + d] = f2bf_bits(val);
        }
    }
}

extern "C" void kernel_launch(void* const* d_in, const int* in_sizes, int n_in,
                              void* d_out, int out_size, void* d_ws, size_t ws_size,
                              hipStream_t stream) {
    const float* hs   = (const float*)d_in[0];
    const float* mask = (const float*)d_in[1];
    const float* Wq = (const float*)d_in[2]; const float* bq = (const float*)d_in[3];
    const float* Wk = (const float*)d_in[4]; const float* bk = (const float*)d_in[5];
    const float* Wv = (const float*)d_in[6]; const float* bv = (const float*)d_in[7];
    const float* Wo = (const float*)d_in[8]; const float* bo = (const float*)d_in[9];
    float* out = (float*)d_out;

    const int M = 4096;              // B*S
    const int Hd = 1024;             // hidden
    const size_t X_ELE = (size_t)M * Hd;      // 4194304
    const size_t W_ELE = (size_t)Hd * Hd;     // 1048576

    unsigned short* ws = (unsigned short*)d_ws;
    unsigned short* Xb  = ws;                 ws += X_ELE;
    unsigned short* Wqb = ws;                 ws += W_ELE;
    unsigned short* Wkb = ws;                 ws += W_ELE;
    unsigned short* Wvb = ws;                 ws += W_ELE;
    unsigned short* Wob = ws;                 ws += W_ELE;
    unsigned short* Qb  = ws;                 ws += X_ELE;   // [b][h][s][d]
    unsigned short* Kb  = ws;                 ws += X_ELE;   // [b][h][s][d]
    unsigned short* Vtb = ws;                 ws += X_ELE;   // [b][h][d][s]
    unsigned short* Cb  = ws;                 ws += X_ELE;   // [b][s][h*64+d]
    // total: ~48 MB of workspace

    // fp32 -> bf16 conversions
    cvt_f32_to_bf16<<<(int)((X_ELE + 255) / 256), 256, 0, stream>>>(hs, Xb, (int)X_ELE);
    cvt_f32_to_bf16<<<(int)((W_ELE + 255) / 256), 256, 0, stream>>>(Wq, Wqb, (int)W_ELE);
    cvt_f32_to_bf16<<<(int)((W_ELE + 255) / 256), 256, 0, stream>>>(Wk, Wkb, (int)W_ELE);
    cvt_f32_to_bf16<<<(int)((W_ELE + 255) / 256), 256, 0, stream>>>(Wv, Wvb, (int)W_ELE);
    cvt_f32_to_bf16<<<(int)((W_ELE + 255) / 256), 256, 0, stream>>>(Wo, Wob, (int)W_ELE);

    // Q/K/V projections (4096x1024x1024 bf16 WMMA GEMMs)
    dim3 gg(Hd / 64, M / 128);   // (16, 32)
    gemm_bf16<<<gg, 256, 0, stream>>>(Xb, Wqb, bq, Qb,  M, Hd, Hd, 0);
    gemm_bf16<<<gg, 256, 0, stream>>>(Xb, Wkb, bk, Kb,  M, Hd, Hd, 1);
    gemm_bf16<<<gg, 256, 0, stream>>>(Xb, Wvb, bv, Vtb, M, Hd, Hd, 2);

    // flash attention over 32 (b,h) pairs, 128 query rows per block
    attn_kernel<<<dim3(2048 / 128, 32), 256, 0, stream>>>(Qb, Kb, Vtb, mask, Cb);

    // output projection -> f32 d_out
    gemm_bf16<<<gg, 256, 0, stream>>>(Cb, Wob, bo, out, M, Hd, Hd, 3);
}
